// ContrastiveHead_2705829396494
// MI455X (gfx1250) — matrix operand order repaired and checked by reading, not compiled, not run on hardware
//
#include <hip/hip_runtime.h>
#include <hip/hip_bf16.h>

typedef __attribute__((ext_vector_type(16))) __bf16 v16bf;
typedef __attribute__((ext_vector_type(8)))  float  v8f;

// ---- problem constants (from reference setup_inputs) ----
#define BS 32
#define IL 12
#define NN 207
#define TK 3
#define SK 3
#define DM 64
#define FD 768                    // F = IL*DM
#define MROWS (BS*NN)             // 6624
#define CHUNK (TK*SK*DM)          // 576 floats: contiguous (tk,sk,dm) block per (b,l,n)
#define XB_STRIDE (IL*NN*CHUNK)   // 1430784
#define XL_STRIDE (NN*CHUNK)      // 119232
#define NNDM (NN*DM)              // 13248
#define LDSTRIDE 40               // padded bf16 row stride (32 cols + 8 pad)

// ---- workspace layout ----
#define WT_BYTES   ((size_t)FD*FD*2)                 // 1,179,648
#define Z_OFFSET   WT_BYTES
#define Z_BYTES    ((size_t)MROWS*FD*4)              // 20,348,928
#define P_OFFSET   (Z_OFFSET + Z_BYTES)              // block partials: 9*256 floats

// ---------------- prep: W^T -> bf16 ----------------
__global__ __launch_bounds__(256) void prep_wt(const float* __restrict__ W,
                                               __bf16* __restrict__ Wt) {
    int idx = blockIdx.x * 256 + threadIdx.x;   // idx = g*768 + f
    int g = idx / FD;
    int f = idx - g * FD;
    Wt[idx] = (__bf16)W[f * FD + g];
}

// ---------------- WMMA fragment loads from LDS ----------------
// A 16x32 bf16 fragment: lane m=lane&15; VGPR v: K=(v>=4?16:0)+(lane>=16?8:0)+2(v%4)+{0,1}
__device__ __forceinline__ v16bf frag_a(const __bf16* rowp, int c0) {
    union { uint4 u[2]; v16bf v; } r;
    const uint4* p = reinterpret_cast<const uint4*>(rowp + c0);
    r.u[0] = p[0];      // K = c0 .. c0+7
    r.u[1] = p[2];      // K = c0+16 .. c0+23
    return r.v;
}
// B 32x16 bf16 fragment: lane n=lane&15; K=(v>=4?16:0)+(lane>=16?8:0)+2(v%4)+{0,1}
__device__ __forceinline__ v16bf frag_b(const __bf16* rowp, int k0) {
    union { uint4 u[2]; v16bf v; } r;
    const uint4* p = reinterpret_cast<const uint4*>(rowp + k0);
    r.u[0] = p[0];
    r.u[1] = p[2];
    return r.v;
}

// ---------------- GEMM: Z = H(i,j) * W  (bf16 inputs, f32 accum) ----------------
// H row r=(b*207+n') , feature f :  u=n'*768+f ->  x[b, l=u/13248, n=(u/64)%207, i, j, d=u%64]
__global__ __launch_bounds__(256) void gemm_hw(const float* __restrict__ x,
                                               const __bf16* __restrict__ Wt,
                                               float* __restrict__ Z,
                                               int vi, int vj) {
    __shared__ __align__(16) __bf16 Alds[128 * LDSTRIDE];
    __shared__ __align__(16) __bf16 Blds[128 * LDSTRIDE];

    const int tid  = threadIdx.x;
    const int m0   = blockIdx.y * 128;
    const int n0   = blockIdx.x * 128;
    const int lane = tid & 31;
    const int wid  = tid >> 5;
    const int wm   = wid & 3;        // 4 waves along M
    const int wn   = wid >> 2;       // 2 waves along N
    const int mrow = lane & 15;
    const int ac0  = (lane & 16) ? 8 : 0;
    const int bk0  = (lane & 16) ? 8 : 0;
    const int sr   = tid >> 1;          // staging row 0..127
    const int sc   = (tid & 1) * 16;    // staging col 0 / 16

    const int r = m0 + sr;
    int bidx = 0, npi = 0;
    if (r < MROWS) { bidx = r / NN; npi = r - bidx * NN; }
    const int vioff = (vi * SK + vj) * DM;

    // Per-thread B staging: async copy global(Wt) -> LDS, 32 bytes as 2x b128.
    // LDS destination is fixed across K-steps; global source advances by 64B.
    const unsigned lds_b = (unsigned)(uintptr_t)(&Blds[sr * LDSTRIDE + sc]);
    const __bf16* wsrc   = Wt + (size_t)(n0 + sr) * FD + sc;

    v8f acc[2][4] = {};

    for (int kt = 0; kt < FD / 32; ++kt) {
        // ---- stage B tile: CDNA5 async global->LDS (ASYNCcnt path, no VGPR bounce) ----
        {
            unsigned long long gaddr = (unsigned long long)(uintptr_t)(wsrc + kt * 32);
            asm volatile(
                "global_load_async_to_lds_b128 %0, %1, off\n\t"
                "global_load_async_to_lds_b128 %0, %1, off offset:16"
                :: "v"(lds_b), "v"(gaddr) : "memory");
        }
        // ---- stage A tile (f32 gather from x -> bf16 LDS, converted in VALU) ----
        {
            __bf16* dst = &Alds[sr * LDSTRIDE + sc];
            if (r < MROWS) {
                int u   = npi * FD + kt * 32 + sc;   // 16-float run stays in one 64-chunk
                int l   = u / NNDM;
                int rem = u - l * NNDM;
                int n   = rem >> 6;
                int d   = rem & 63;
                const float4* src = reinterpret_cast<const float4*>(
                    x + (size_t)bidx * XB_STRIDE + (size_t)l * XL_STRIDE +
                        (size_t)n * CHUNK + vioff + d);
#pragma unroll
                for (int q = 0; q < 4; ++q) {
                    float4 v = src[q];
                    dst[q * 4 + 0] = (__bf16)v.x;
                    dst[q * 4 + 1] = (__bf16)v.y;
                    dst[q * 4 + 2] = (__bf16)v.z;
                    dst[q * 4 + 3] = (__bf16)v.w;
                }
            } else {
#pragma unroll
                for (int q = 0; q < 16; ++q) dst[q] = (__bf16)0.0f;
            }
        }
        // async writes to LDS are not tracked by the compiler's DScnt logic:
        // drain ASYNCcnt explicitly before the workgroup barrier.
        asm volatile("s_wait_asynccnt 0x0" ::: "memory");
        __syncthreads();

        v16bf a0 = frag_a(&Alds[(wm * 32 + mrow) * LDSTRIDE], ac0);
        v16bf a1 = frag_a(&Alds[(wm * 32 + 16 + mrow) * LDSTRIDE], ac0);
#pragma unroll
        for (int nf = 0; nf < 4; ++nf) {
            v16bf bfr = frag_b(&Blds[(wn * 64 + nf * 16 + mrow) * LDSTRIDE], bk0);
            acc[0][nf] = __builtin_amdgcn_wmma_f32_16x16x32_bf16(
                false, a0, false, bfr, (short)0, acc[0][nf], false, false);
            acc[1][nf] = __builtin_amdgcn_wmma_f32_16x16x32_bf16(
                false, a1, false, bfr, (short)0, acc[1][nf], false, false);
        }
        __syncthreads();
    }

    // ---- store C/D: VGPR q -> row q + (lane>=16 ? 8 : 0); col = lane&15 ----
    const int rbase = m0 + wm * 32 + ((lane & 16) ? 8 : 0);
    const int cbase = n0 + wn * 64 + (lane & 15);
#pragma unroll
    for (int mi = 0; mi < 2; ++mi) {
#pragma unroll
        for (int nf = 0; nf < 4; ++nf) {
#pragma unroll
            for (int q = 0; q < 8; ++q) {
                int row = rbase + mi * 16 + q;
                if (row < MROWS) Z[(size_t)row * FD + cbase + nf * 16] = acc[mi][nf][q];
            }
        }
    }
}

// ---------------- pair dot products + stable BCE terms ----------------
// block = one pred row a (0..127 pos, 128..255 neg); 8 waves, wave-per-node.
__global__ __launch_bounds__(256) void pair_reduce(const float* __restrict__ x,
                                                   const float* __restrict__ Z,
                                                   const float* __restrict__ bvec,
                                                   float* __restrict__ P,
                                                   int vi, int vj,
                                                   int t0, int t1, int s0, int s1) {
    const int a   = blockIdx.x;
    const bool ng = a >= 128;
    const int aa  = a & 127;
    const int ba  = aa & 31;   // anchor batch: tile(h,...) -> a % bs
    const int vb  = aa >> 2;   // view batch  : reshape    -> a / 4
    const int p   = aa & 3;
    int t, s;
    if (!ng) {
        if (p < 2) { t = p ? t1 : t0; s = vj; }
        else       { t = vi; s = (p == 2) ? s0 : s1; }
    } else {
        t = (p >> 1) ? t1 : t0;
        s = (p & 1) ? s1 : s0;
    }
    const float bias  = bvec[0];
    const int  lane   = threadIdx.x & 31;
    const int  w      = threadIdx.x >> 5;
    const int  vsoff  = (t * SK + s) * DM;
    float wacc = 0.0f;

    for (int np = w; np < NN; np += 8) {
        const float* zrow = Z + (size_t)(ba * NN + np) * FD;
        const int ub = np * FD;
        float dot = 0.0f;
        for (int f = lane; f < FD; f += 32) {
            int u   = ub + f;
            int l   = u / NNDM;
            int rem = u - l * NNDM;
            int n   = rem >> 6;
            int d   = rem & 63;
            float c = x[(size_t)vb * XB_STRIDE + (size_t)l * XL_STRIDE +
                        (size_t)n * CHUNK + vsoff + d];
            dot = fmaf(zrow[f], c, dot);
        }
#pragma unroll
        for (int m = 16; m; m >>= 1) dot += __shfl_xor(dot, m, 32);
        if (lane == 0) {
            float v    = dot + bias;
            float sp   = log1pf(__expf(-fabsf(v)));   // stable softplus core
            float term = ng ? (fmaxf(v, 0.0f) + sp)   // label 0: softplus(v)
                            : (fmaxf(-v, 0.0f) + sp); // label 1: softplus(-v)
            wacc += term;
        }
    }

    __shared__ float ws8[8];
    if (lane == 0) ws8[w] = wacc;
    __syncthreads();
    if (threadIdx.x == 0) {
        float sblk = 0.0f;
#pragma unroll
        for (int q = 0; q < 8; ++q) sblk += ws8[q];
        P[blockIdx.x] = sblk;   // P already offset by combo
    }
}

// ---------------- deterministic final reduction ----------------
__global__ __launch_bounds__(256) void finish_sum(const float* __restrict__ P,
                                                  float* __restrict__ out,
                                                  float scale) {
    __shared__ float sh[256];
    float acc = 0.0f;
    for (int idx = threadIdx.x; idx < 9 * 256; idx += 256) acc += P[idx];
    sh[threadIdx.x] = acc;
    __syncthreads();
    for (int off = 128; off; off >>= 1) {
        if (threadIdx.x < off) sh[threadIdx.x] += sh[threadIdx.x + off];
        __syncthreads();
    }
    if (threadIdx.x == 0) out[0] = sh[0] * scale;
}

extern "C" void kernel_launch(void* const* d_in, const int* in_sizes, int n_in,
                              void* d_out, int out_size, void* d_ws, size_t ws_size,
                              hipStream_t stream) {
    const float* x = (const float*)d_in[0];
    const float* W = (const float*)d_in[1];
    const float* b = (const float*)d_in[2];
    float* out = (float*)d_out;

    __bf16* Wt = (__bf16*)d_ws;
    float*  Z  = (float*)((char*)d_ws + Z_OFFSET);
    float*  P  = (float*)((char*)d_ws + P_OFFSET);

    prep_wt<<<(FD * FD) / 256, 256, 0, stream>>>(W, Wt);

    for (int c = 0; c < 9; ++c) {
        int i = c / 3, j = c % 3;
        int t0 = (i == 0) ? 1 : 0, t1 = (i == 2) ? 1 : 2;
        int s0 = (j == 0) ? 1 : 0, s1 = (j == 2) ? 1 : 2;
        gemm_hw<<<dim3(FD / 128, (MROWS + 127) / 128), 256, 0, stream>>>(x, Wt, Z, i, j);
        pair_reduce<<<256, 256, 0, stream>>>(x, Z, b, P + c * 256, i, j, t0, t1, s0, s1);
    }

    const float scale = 1.0f / (256.0f * (float)NN * 9.0f);
    finish_sum<<<1, 256, 0, stream>>>(P, out, scale);
}
